// LIGHT_MOTION_ReplaceBlock_33217277067945
// MI455X (gfx1250) — compile-verified, matrix operand order checked by reading, not run on hardware
//
#include <hip/hip_runtime.h>

// ---------------- problem constants (from reference) ----------------
#define NT    64          // nt
#define CIN   256         // C
#define HH    56
#define WW    56
#define HWSZ  3136        // 56*56
#define FOLD  32
#define C2    64          // 2*fold
#define NSEG  8
#define BN_EPS 1e-5f

typedef __attribute__((ext_vector_type(8)))  float  v8f;
typedef __attribute__((ext_vector_type(16))) __bf16 v16bf;
typedef __attribute__((ext_vector_type(4)))  float  v4f;
typedef __attribute__((ext_vector_type(4)))  int    v4i;

#define AS1 __attribute__((address_space(1)))
#define AS3 __attribute__((address_space(3)))

#if defined(__AMDGCN__) && __has_builtin(__builtin_amdgcn_global_load_async_to_lds_b128) && __has_builtin(__builtin_amdgcn_s_wait_asynccnt)
#define USE_ASYNC_LDS 1
#endif

// =====================================================================
// Kernel 1: 1x1 conv as bf16 WMMA GEMM with async-LDS double buffering.
//   Z[m, n] = sum_k X[m, k] * W1[n, k] + b1[n]
//   m = image-local spatial index (64 per block), n = 0..63, k = 0..255.
// Block: 128 threads (4 waves). Each wave owns a 16-row M chunk and all
// four 16-wide N tiles. grid = (HWSZ/64, NT).
// K-loop tiles are DMA'd with GLOBAL_LOAD_ASYNC_TO_LDS_B128 into a
// 2-deep LDS pipeline (ASYNCcnt-tracked), fp32 in LDS, cvt->bf16 at
// fragment build.
// =====================================================================
__global__ __launch_bounds__(128) void conv1x1_wmma(
    const float* __restrict__ x, const float* __restrict__ w1,
    const float* __restrict__ b1, float* __restrict__ zbuf)
{
    const int n    = blockIdx.y;
    const int hw0  = blockIdx.x * 64;
    const int tid  = threadIdx.x;
    const int lane = tid & 31;
    const int wv   = tid >> 5;

    const int mrow  = lane & 15;
    const int half  = lane >> 4;
    const int kbase = half * 8;
    const int mloc  = wv * 16 + mrow;

    __shared__ float xs[2][32 * 64];     // [k][m] fp32, 8 KB each
    __shared__ float wlds[2][64 * 32];   // [n][k] fp32, 8 KB each
    __shared__ float ots[64 * 64];       // [ch][m] 16 KB (store staging)

    v8f acc[4] = {};

#ifdef USE_ASYNC_LDS
    // Each thread issues 4 + 4 async b128 transfers per K step.
    auto issue_tile = [&](int k0, int buf) {
        #pragma unroll
        for (int j = 0; j < 4; ++j) {
            int i4 = tid + j * 128;               // 0..511 float4 slots
            // X tile: [kk][mm] (both contiguous along mm -> b128)
            int kk  = i4 >> 4;
            int mm4 = (i4 & 15) * 4;
            const float* g = &x[((size_t)n * CIN + k0 + kk) * HWSZ + hw0 + mm4];
            __builtin_amdgcn_global_load_async_to_lds_b128(
                (AS1 v4i*)(void*)g, (AS3 v4i*)(void*)&xs[buf][kk * 64 + mm4], 0, 0);
            // W tile: [nn][kk] (contiguous along k -> b128)
            int nn  = i4 >> 3;
            int kk4 = (i4 & 7) * 4;
            const float* gw = &w1[nn * CIN + k0 + kk4];
            __builtin_amdgcn_global_load_async_to_lds_b128(
                (AS1 v4i*)(void*)gw, (AS3 v4i*)(void*)&wlds[buf][nn * 32 + kk4], 0, 0);
        }
    };
    int rb = 0;
    issue_tile(0, 0);
#endif

    for (int k0 = 0; k0 < CIN; k0 += 32) {
#ifdef USE_ASYNC_LDS
        __builtin_amdgcn_s_wait_asynccnt(0);
        __syncthreads();
        if (k0 + 32 < CIN) issue_tile(k0 + 32, rb ^ 1);
#else
        const int rb = 0;
        __syncthreads();
        for (int i = tid; i < 64 * 32; i += 128) {
            int kk = i >> 6, mm = i & 63;
            xs[0][kk * 64 + mm] = x[((size_t)n * CIN + k0 + kk) * HWSZ + hw0 + mm];
        }
        for (int i = tid; i < 64 * 32; i += 128) {
            int nn = i >> 5, kk = i & 31;
            wlds[0][nn * 32 + kk] = w1[nn * CIN + k0 + kk];
        }
        __syncthreads();
#endif
        // ---- A fragment: row (lane&15) of this wave's M chunk ----
        v16bf a;
        #pragma unroll
        for (int i = 0; i < 8; ++i) {
            a[i]     = (__bf16)xs[rb][(kbase + i) * 64 + mloc];
            a[8 + i] = (__bf16)xs[rb][(16 + kbase + i) * 64 + mloc];
        }
        #pragma unroll
        for (int jt = 0; jt < 4; ++jt) {
            v16bf b;
            const float* bp = &wlds[rb][(jt * 16 + mrow) * 32];
            #pragma unroll
            for (int i = 0; i < 8; ++i) {
                b[i]     = (__bf16)bp[kbase + i];
                b[8 + i] = (__bf16)bp[16 + kbase + i];
            }
            acc[jt] = __builtin_amdgcn_wmma_f32_16x16x32_bf16(
                false, a, false, b, (short)0, acc[jt], false, false);
        }
#ifdef USE_ASYNC_LDS
        rb ^= 1;
#endif
    }

    __syncthreads();
    // ---- stage D through LDS for coalesced channel-major stores ----
    #pragma unroll
    for (int jt = 0; jt < 4; ++jt) {
        int ch = jt * 16 + mrow;
        #pragma unroll
        for (int r = 0; r < 8; ++r)
            ots[ch * 64 + wv * 16 + r + 8 * half] = acc[jt][r];
    }
    __syncthreads();
    for (int i = tid; i < 64 * 64; i += 128) {
        int ch = i >> 6, mm = i & 63;
        zbuf[((size_t)n * C2 + ch) * HWSZ + hw0 + mm] = ots[i] + b1[ch];
    }
}

// =====================================================================
// Kernel 2: BN statistics. One block per channel (64 blocks x 256 thr).
// z stays L2-resident (51 MB < 192 MB), so this re-read is cheap.
// =====================================================================
__global__ __launch_bounds__(256) void bn_stats(
    const float* __restrict__ zbuf, float* __restrict__ meanp,
    float* __restrict__ rstdp)
{
    const int c   = blockIdx.x;
    const int tid = threadIdx.x;
    float s = 0.f, sq = 0.f;
    for (int n = 0; n < NT; ++n) {
        const float* zp = zbuf + ((size_t)n * C2 + c) * HWSZ;
        for (int i = tid; i < HWSZ; i += 256) {
            float v = zp[i];
            s += v; sq += v * v;
        }
    }
    __shared__ float rs[256], rq[256];
    rs[tid] = s; rq[tid] = sq;
    __syncthreads();
    for (int off = 128; off > 0; off >>= 1) {
        if (tid < off) { rs[tid] += rs[tid + off]; rq[tid] += rq[tid + off]; }
        __syncthreads();
    }
    if (tid == 0) {
        const float cnt = (float)(NT * HWSZ);
        float m   = rs[0] / cnt;
        float var = rq[0] / cnt - m * m;
        meanp[c] = m;
        rstdp[c] = rsqrtf(var + BN_EPS);
    }
}

__device__ __forceinline__ float bn_relu(float v, float m, float r, float g, float b) {
    return fmaxf(0.f, (v - m) * r * g + b);
}

// =====================================================================
// Kernel 3: 3x3 convs (nxt on za, lst on zb) as 9-tap bf16 WMMA implicit
// GEMM + temporal-shift combine. Block: one image n, 64 hw positions.
// grid = (HWSZ/64, NT), 128 threads (4 waves x 16-row M chunks).
// =====================================================================
__global__ __launch_bounds__(128) void conv3x3_tsm(
    const float* __restrict__ zbuf, const float* __restrict__ meanp,
    const float* __restrict__ rstdp, const float* __restrict__ gamma,
    const float* __restrict__ beta,
    const float* __restrict__ wn, const float* __restrict__ bnx,
    const float* __restrict__ wl, const float* __restrict__ blx,
    float* __restrict__ out)
{
    const int n    = blockIdx.y;
    const int hw0  = blockIdx.x * 64;
    const int tid  = threadIdx.x;
    const int lane = tid & 31;
    const int wv   = tid >> 5;
    const int seg  = n & (NSEG - 1);

    // LDS: [phase A] zs(24576B) + wn_s(18432B) + wl_s(18432B) = 61440B
    //      [phase B] ots (8192B) aliases the zs region.
    __shared__ __align__(16) unsigned char smem[61440];
    __bf16* zs   = (__bf16*)smem;              // [pos 0..191][ch 0..63]
    __bf16* wn_s = (__bf16*)(smem + 24576);    // [tap][oc][ic]
    __bf16* wl_s = (__bf16*)(smem + 24576 + 18432);
    float*  ots  = (float*)smem;               // [oc 0..31][m 0..63]

    // ---- stage normalized za/zb with halo: pos = hw0-64 .. hw0+127 ----
    for (int i = tid; i < 64 * 192; i += 128) {
        int ch = i / 192;
        int P  = i - ch * 192;
        int gp = hw0 - 64 + P;
        float v = 0.f;
        if (gp >= 0 && gp < HWSZ) {
            float zv = zbuf[((size_t)n * C2 + ch) * HWSZ + gp];
            v = bn_relu(zv, meanp[ch], rstdp[ch], gamma[ch], beta[ch]);
        }
        zs[P * 64 + ch] = (__bf16)v;
    }
    // ---- stage weights as [tap][oc][ic] bf16 ----
    for (int i = tid; i < 9 * 32 * 32; i += 128) {
        int oc = i / 288;
        int r  = i - oc * 288;
        int ic = r / 9;
        int tap = r - ic * 9;
        wn_s[(tap * 32 + oc) * 32 + ic] = (__bf16)wn[i];
        wl_s[(tap * 32 + oc) * 32 + ic] = (__bf16)wl[i];
    }
    __syncthreads();

    const int mrow  = lane & 15;
    const int half  = lane >> 4;
    const int kbase = half * 8;
    const int m_g   = hw0 + wv * 16 + mrow;      // this lane's A row (global hw)
    const int row   = m_g / WW;
    const int col   = m_g - row * WW;

    v8f an[2] = {};   // nxt accumulators, N tiles 0..1 (oc 0..31)
    v8f al[2] = {};   // lst accumulators

    #pragma unroll
    for (int dy = -1; dy <= 1; ++dy) {
        #pragma unroll
        for (int dx = -1; dx <= 1; ++dx) {
            const int tap = (dy + 1) * 3 + (dx + 1);
            const int rr = row + dy, cc = col + dx;
            const bool valid = (rr >= 0) && (rr < HH) && (cc >= 0) && (cc < WW);
            const int P = (m_g + dx + WW * dy) - (hw0 - 64);

            v16bf aA, aB;
            #pragma unroll
            for (int i = 0; i < 16; ++i) { aA[i] = (__bf16)0.f; aB[i] = (__bf16)0.f; }
            if (valid) {
                const __bf16* zp = &zs[P * 64];
                #pragma unroll
                for (int i = 0; i < 8; ++i) {
                    aA[i]     = zp[kbase + i];           // za: ic 0..31
                    aA[8 + i] = zp[16 + kbase + i];
                    aB[i]     = zp[32 + kbase + i];      // zb: ic 0..31 (+32)
                    aB[8 + i] = zp[48 + kbase + i];
                }
            }
            #pragma unroll
            for (int jt = 0; jt < 2; ++jt) {
                const int oc = jt * 16 + mrow;
                v16bf bn_f, bl_f;
                const __bf16* bpn = &wn_s[(tap * 32 + oc) * 32];
                const __bf16* bpl = &wl_s[(tap * 32 + oc) * 32];
                #pragma unroll
                for (int i = 0; i < 8; ++i) {
                    bn_f[i]     = bpn[kbase + i];
                    bn_f[8 + i] = bpn[16 + kbase + i];
                    bl_f[i]     = bpl[kbase + i];
                    bl_f[8 + i] = bpl[16 + kbase + i];
                }
                an[jt] = __builtin_amdgcn_wmma_f32_16x16x32_bf16(
                    false, aA, false, bn_f, (short)0, an[jt], false, false);
                al[jt] = __builtin_amdgcn_wmma_f32_16x16x32_bf16(
                    false, aB, false, bl_f, (short)0, al[jt], false, false);
            }
        }
    }

    // ---------- part1: out[n-1, 0..31] = nxt_n - za_{n-1}; zeros at seg==7 ----------
    __syncthreads();          // all zs reads done; safe to alias with ots
    #pragma unroll
    for (int jt = 0; jt < 2; ++jt) {
        int oc = jt * 16 + mrow;
        #pragma unroll
        for (int r = 0; r < 8; ++r)
            ots[oc * 64 + wv * 16 + r + 8 * half] = an[jt][r] + bnx[oc];
    }
    __syncthreads();
    for (int i = tid; i < 32 * 64; i += 128) {
        int oc = i >> 6, mm = i & 63;
        int ghw = hw0 + mm;
        if (seg > 0) {
            float zp = zbuf[((size_t)(n - 1) * C2 + oc) * HWSZ + ghw];
            float za = bn_relu(zp, meanp[oc], rstdp[oc], gamma[oc], beta[oc]);
            out[((size_t)(n - 1) * CIN + oc) * HWSZ + ghw] = ots[i] - za;
        }
        if (seg == NSEG - 1)
            out[((size_t)n * CIN + oc) * HWSZ + ghw] = 0.f;
    }

    // ---------- part2: out[n+1, 32..63] = zb_{n+1} - lst_n; zeros at seg==0 ----------
    __syncthreads();
    #pragma unroll
    for (int jt = 0; jt < 2; ++jt) {
        int oc = jt * 16 + mrow;
        #pragma unroll
        for (int r = 0; r < 8; ++r)
            ots[oc * 64 + wv * 16 + r + 8 * half] = al[jt][r] + blx[oc];
    }
    __syncthreads();
    for (int i = tid; i < 32 * 64; i += 128) {
        int oc = i >> 6, mm = i & 63;
        int ghw = hw0 + mm;
        int ch  = 32 + oc;
        if (seg < NSEG - 1) {
            float zp = zbuf[((size_t)(n + 1) * C2 + ch) * HWSZ + ghw];
            float zb = bn_relu(zp, meanp[ch], rstdp[ch], gamma[ch], beta[ch]);
            out[((size_t)(n + 1) * CIN + ch) * HWSZ + ghw] = zb - ots[i];
        }
        if (seg == 0)
            out[((size_t)n * CIN + ch) * HWSZ + ghw] = 0.f;
    }
}

// =====================================================================
// Kernel 4: identity passthrough of channels 64..255 (154 MB each way).
// Non-temporal float4 streaming so it does not evict z from L2.
// 64 images * 192*3136/4 = 150528 float4 per image.
// =====================================================================
__global__ __launch_bounds__(256) void ident_copy(
    const float* __restrict__ x, float* __restrict__ out)
{
    const size_t PER_IMG = (size_t)192 * HWSZ / 4;       // 150528
    size_t idx = (size_t)blockIdx.x * 256 + threadIdx.x; // < 64 * 150528
    size_t n = idx / PER_IMG;
    size_t r = idx - n * PER_IMG;
    const v4f* src = (const v4f*)(x   + ((size_t)n * CIN + C2) * HWSZ) + r;
    v4f*       dst = (v4f*)      (out + ((size_t)n * CIN + C2) * HWSZ) + r;
    __builtin_prefetch((const void*)(src + 256), 0, 0);  // global_prefetch_b8
    v4f v = __builtin_nontemporal_load(src);
    __builtin_nontemporal_store(v, dst);
}

// =====================================================================
extern "C" void kernel_launch(void* const* d_in, const int* in_sizes, int n_in,
                              void* d_out, int out_size, void* d_ws, size_t ws_size,
                              hipStream_t stream)
{
    const float* x     = (const float*)d_in[0];
    const float* w1    = (const float*)d_in[1];
    const float* b1    = (const float*)d_in[2];
    const float* gamma = (const float*)d_in[3];
    const float* beta  = (const float*)d_in[4];
    const float* wnx   = (const float*)d_in[5];
    const float* bnx   = (const float*)d_in[6];
    const float* wlx   = (const float*)d_in[7];
    const float* blx   = (const float*)d_in[8];
    float* out = (float*)d_out;

    // workspace: z (NT*C2*HW floats = 51.4 MB) + mean[64] + rstd[64]
    float* zbuf  = (float*)d_ws;
    float* meanp = zbuf + (size_t)NT * C2 * HWSZ;
    float* rstdp = meanp + 64;

    conv1x1_wmma<<<dim3(HWSZ / 64, NT), 128, 0, stream>>>(x, w1, b1, zbuf);
    bn_stats    <<<C2, 256, 0, stream>>>(zbuf, meanp, rstdp);
    conv3x3_tsm <<<dim3(HWSZ / 64, NT), 128, 0, stream>>>(
        zbuf, meanp, rstdp, gamma, beta, wnx, bnx, wlx, blx, out);

    const size_t n4 = (size_t)NT * 192 * HWSZ / 4;       // 9,633,792
    ident_copy  <<<(unsigned)(n4 / 256), 256, 0, stream>>>(x, out);
}
